// GCN_12987981103641
// MI455X (gfx1250) — matrix operand order repaired and checked by reading, not compiled, not run on hardware
//
#include <hip/hip_runtime.h>
#include <hip/hip_bf16.h>
#include <math.h>

// ---------------------------------------------------------------------------
// GCN forward on MI455X (gfx1250, wave32, WMMA).
// conv2 (94 GFLOP) and all dense graph GEMMs run on v_wmma_f32_16x16x32_f16.
// Edge aggregations use global f32 atomics; pool-max uses uint atomicMax on
// non-negative float bits. __launch_bounds__(256,1) on WMMA kernels removes
// accumulator spills (13 x v8f acc = 104 VGPRs must stay resident).
// ---------------------------------------------------------------------------

typedef _Float16 h16;
typedef __attribute__((ext_vector_type(16))) _Float16 v16h;
typedef __attribute__((ext_vector_type(8)))  float    v8f;

#define NNODES 2048
#define NEDGES 65536
#define HID    128

// activation codes
#define ACT_NONE       0
#define ACT_LRELU      1
#define ACT_TANH       2
#define ACT_RELU       3
#define ACT_TANH2      4   // tanh(tanh(x))
#define ACT_TANH_LRELU 5   // tanh(leaky_relu(x))
#define ACT_SIG_LRELU  6   // sigmoid(leaky_relu(x))

__device__ __forceinline__ float act_apply(float v, int act) {
  switch (act) {
    case ACT_LRELU:      return v > 0.f ? v : 0.01f * v;
    case ACT_TANH:       return tanhf(v);
    case ACT_RELU:       return v > 0.f ? v : 0.f;
    case ACT_TANH2:      return tanhf(tanhf(v));
    case ACT_TANH_LRELU: v = v > 0.f ? v : 0.01f * v; return tanhf(v);
    case ACT_SIG_LRELU:  v = v > 0.f ? v : 0.01f * v; return 1.f / (1.f + expf(-v));
    default:             return v;
  }
}

// k-offset inside a 16x32 f16 A-fragment (contiguous order, matches B packing):
// lanes 0-15 hold M=lane, K 0-15 (hs=0); lanes 16-31 hold same M, K 16-31 (hs=1).
__device__ __forceinline__ int a_koff(int e, int hs) {
  return hs * 16 + e;
}

// ============================ encoder ======================================

// conv1: in [2048,1,100,5] f32, W [100,1,3,3] -> leaky_relu -> h1 f16 [2048,100,98,3]
__global__ void conv1_k(const float* __restrict__ in, const float* __restrict__ W,
                        const float* __restrict__ b, h16* __restrict__ h1) {
  size_t idx = (size_t)blockIdx.x * blockDim.x + threadIdx.x;
  const size_t total = (size_t)NNODES * 100 * 98 * 3;
  if (idx >= total) return;
  int ow = idx % 3;
  int oh = (idx / 3) % 98;
  int co = (idx / 294) % 100;
  int n  = idx / 29400;
  const float* ip = in + (size_t)n * 500;
  const float* wp = W + co * 9;
  float acc = b[co];
  #pragma unroll
  for (int kh = 0; kh < 3; ++kh)
    #pragma unroll
    for (int kw = 0; kw < 3; ++kw)
      acc += ip[(oh + kh) * 5 + ow + kw] * wp[kh * 3 + kw];
  acc = acc > 0.f ? acc : 0.01f * acc;
  h1[idx] = (h16)acc;
}

// pack We2 [200,100,3,2] f32 into WMMA B-fragment layout [19 kt][13 jt][32 lane][16 e]
// B[k][col]: k = ci*6 + kh*2 + kw (K=600 pad 608), col = oc (200 pad 208).
__global__ void pack_We2_k(const float* __restrict__ We2, h16* __restrict__ Bf) {
  int idx = blockIdx.x * blockDim.x + threadIdx.x;
  const int total = 19 * 13 * 512;
  if (idx >= total) return;
  int e = idx & 15, lane = (idx >> 4) & 31, t = idx >> 9;
  int jt = t % 13, kt = t / 13;
  int g = lane >> 4, nn = lane & 15;
  int k = kt * 32 + g * 16 + e, col = jt * 16 + nn;
  float v = 0.f;
  if (k < 600 && col < 200) {
    int ci = k / 6, r = k - ci * 6;
    v = We2[((col * 100 + ci) * 3 + (r >> 1)) * 2 + (r & 1)];
  }
  Bf[idx] = (h16)v;
}

// conv2 implicit-GEMM WMMA: h1 [2048,100,98,3] f16  x  B2frag -> leaky -> h2 f16 [2048,200,96,2]
// M = 2048*192, K = 600 (19 k-steps), N = 200 (13 tiles). One wave per M-tile.
__global__ void __launch_bounds__(256, 1)
conv2_wmma_k(const h16* __restrict__ h1, const h16* __restrict__ Bf,
             const float* __restrict__ be2, h16* __restrict__ h2) {
  const int NT = 13;
  int wave = threadIdx.x >> 5, lane = threadIdx.x & 31;
  int mtile = blockIdx.x * (blockDim.x >> 5) + wave;
  int mloc = lane & 15, hs = lane >> 4;
  int row = mtile * 16 + mloc;
  int node = row / 192, p = row - node * 192;
  int oh = p >> 1, ow = p & 1;
  size_t nbase = (size_t)node * 100 * 98 * 3;

  int off[16];
  #pragma unroll
  for (int e = 0; e < 16; ++e) off[e] = a_koff(e, hs);

  const v8f vz = {0.f, 0.f, 0.f, 0.f, 0.f, 0.f, 0.f, 0.f};
  v8f acc[NT];
  #pragma unroll
  for (int j = 0; j < NT; ++j) acc[j] = vz;

  for (int kt = 0; kt < 19; ++kt) {
    v16h a;
    #pragma unroll
    for (int e = 0; e < 16; ++e) {
      int k = kt * 32 + off[e];
      if (k < 600) {
        int ci = k / 6, r6 = k - ci * 6;
        a[e] = h1[nbase + ((size_t)ci * 98 + (oh + (r6 >> 1))) * 3 + ow + (r6 & 1)];
      } else {
        a[e] = (h16)0.f;
      }
    }
    const h16* bp = Bf + ((size_t)(kt * NT) * 32 + lane) * 16;
    #pragma unroll
    for (int j = 0; j < NT; ++j) {
      v16h bfrag = *(const v16h*)(bp + (size_t)j * 32 * 16);
      acc[j] = __builtin_amdgcn_wmma_f32_16x16x32_f16(
          false, a, false, bfrag, (short)0, acc[j], false, false);
    }
  }

  int ncol = lane & 15;
  #pragma unroll
  for (int j = 0; j < NT; ++j) {
    int col = j * 16 + ncol;           // output channel
    if (col >= 200) continue;
    float bv = be2[col];
    #pragma unroll
    for (int r = 0; r < 8; ++r) {
      int crow = mtile * 16 + hs * 8 + r;
      int cn = crow / 192, cp = crow - cn * 192;
      int coh = cp >> 1, cow = cp & 1;
      float v = acc[j][r] + bv;
      v = v > 0.f ? v : 0.01f * v;
      h2[(((size_t)cn * 200 + col) * 96 + coh) * 2 + cow] = (h16)v;
    }
  }
}

// conv3: h2 [2048,200,96,2] f16, We3 [1,200,3,2] -> tanh -> concat1 cols 0..93 (ld 192)
__global__ void conv3_k(const h16* __restrict__ h2, const float* __restrict__ We3,
                        const float* __restrict__ be3, h16* __restrict__ concat1) {
  int idx = blockIdx.x * blockDim.x + threadIdx.x;
  if (idx >= NNODES * 96) return;
  int p = idx % 96, n = idx / 96;
  if (p >= 94) { concat1[(size_t)n * 192 + p] = (h16)0.f; return; }
  float acc = be3[0];
  for (int ci = 0; ci < 200; ++ci) {
    size_t base = ((size_t)n * 200 + ci) * 96;
    const float* wp = We3 + ci * 6;
    #pragma unroll
    for (int kh = 0; kh < 3; ++kh) {
      acc += (float)h2[(base + p + kh) * 2 + 0] * wp[kh * 2 + 0];
      acc += (float)h2[(base + p + kh) * 2 + 1] * wp[kh * 2 + 1];
    }
  }
  concat1[(size_t)n * 192 + p] = (h16)tanhf(acc);
}

// ============================ graph helpers ================================

__global__ void deg_add_k(const int* __restrict__ dst, float* __restrict__ deg) {
  int e = blockIdx.x * blockDim.x + threadIdx.x;
  if (e < NEDGES) atomicAdd(&deg[dst[e]], 1.0f);
}

__global__ void norm_k(const float* __restrict__ deg, float* __restrict__ norm) {
  int n = blockIdx.x * blockDim.x + threadIdx.x;
  if (n < NNODES) norm[n] = rsqrtf(fmaxf(deg[n], 1.0f));
}

// acc[dst] += x[src]*norm[src], d == 128
__global__ void scatter_sum_k(const float* __restrict__ x, const int* __restrict__ src,
                              const int* __restrict__ dst, const float* __restrict__ norm,
                              float* __restrict__ acc) {
  size_t idx = (size_t)blockIdx.x * blockDim.x + threadIdx.x;
  if (idx >= (size_t)NEDGES * 128) return;
  int e = idx >> 7, c = idx & 127;
  int s = src[e];
  atomicAdd(&acc[(size_t)dst[e] * 128 + c], x[(size_t)s * 128 + c] * norm[s]);
}

// hn[dst] = max(hn[dst], m[src]); m >= 0 so uint-bit atomicMax is order-correct.
__global__ void scatter_max_k(const float* __restrict__ m, const int* __restrict__ src,
                              const int* __restrict__ dst, float* __restrict__ hn, int d) {
  size_t idx = (size_t)blockIdx.x * blockDim.x + threadIdx.x;
  if (idx >= (size_t)NEDGES * d) return;
  int e = idx / d, c = idx - (size_t)e * d;
  float v = m[(size_t)src[e] * d + c];
  atomicMax((unsigned int*)&hn[(size_t)dst[e] * d + c], __float_as_uint(v));
}

// y = alpha*acc*norm[n] + beta*z ; also write f16 copy into concat slot (ldh)
__global__ void finalize_k(const float* __restrict__ acc, const float* __restrict__ norm,
                           float alpha, const float* __restrict__ z, float beta,
                           float* __restrict__ y, h16* __restrict__ yh, int ldh) {
  size_t idx = (size_t)blockIdx.x * blockDim.x + threadIdx.x;
  if (idx >= (size_t)NNODES * 128) return;
  int n = idx >> 7, c = idx & 127;
  float v = alpha * acc[idx] * norm[n];
  if (z) v += beta * z[idx];
  y[idx] = v;
  yh[(size_t)n * ldh + c] = (h16)v;
}

__global__ void cvt_f16_k(const float* __restrict__ x, int rows, int cols,
                          h16* __restrict__ dstp, int ldd) {
  size_t idx = (size_t)blockIdx.x * blockDim.x + threadIdx.x;
  if (idx >= (size_t)rows * cols) return;
  int r = idx / cols, c = idx - (size_t)r * cols;
  dstp[(size_t)r * ldd + c] = (h16)x[idx];
}

// ===================== generic weight pack (fragment layout) ===============

__global__ void copy_pad_k(const float* __restrict__ src, int rows, int cols,
                           h16* __restrict__ dstp, int ldd, int row0) {
  size_t idx = (size_t)blockIdx.x * blockDim.x + threadIdx.x;
  if (idx >= (size_t)rows * cols) return;
  int r = idx / cols, c = idx - (size_t)r * cols;
  dstp[(size_t)(row0 + r) * ldd + c] = (h16)src[idx];
}

// row-major f16 [ktiles*32 x ntiles*16] -> per-lane B fragments
__global__ void pack_frags_k(const h16* __restrict__ Btmp, int ldb, int ktiles,
                             int ntiles, h16* __restrict__ Bf) {
  int idx = blockIdx.x * blockDim.x + threadIdx.x;
  int total = ktiles * ntiles * 512;
  if (idx >= total) return;
  int e = idx & 15, lane = (idx >> 4) & 31, t = idx >> 9;
  int jt = t % ntiles, kt = t / ntiles;
  int g = lane >> 4, nn = lane & 15;
  int k = kt * 32 + g * 16 + e, col = jt * 16 + nn;
  Bf[idx] = Btmp[(size_t)k * ldb + col];
}

// ===================== generic WMMA GEMM (graph layers) ====================
// C = act(A(f16,row-major,lda) x Bfrag + bias), optional f16 copy at ldh.
template <int NT>
__global__ void __launch_bounds__(256, 1)
wmma_gemm_k(const h16* __restrict__ A, int lda, int ktiles,
            const h16* __restrict__ Bf,
            const float* __restrict__ bias, int Nc,
            float* __restrict__ C, int ldc,
            h16* __restrict__ Ch, int ldh, int act) {
  int wave = threadIdx.x >> 5, lane = threadIdx.x & 31;
  int mtile = blockIdx.x * (blockDim.x >> 5) + wave;
  int mloc = lane & 15, hs = lane >> 4;
  const h16* arow = A + (size_t)(mtile * 16 + mloc) * lda;

  int off[16];
  #pragma unroll
  for (int e = 0; e < 16; ++e) off[e] = a_koff(e, hs);

  const v8f vz = {0.f, 0.f, 0.f, 0.f, 0.f, 0.f, 0.f, 0.f};
  v8f acc[NT];
  #pragma unroll
  for (int j = 0; j < NT; ++j) acc[j] = vz;

  for (int kt = 0; kt < ktiles; ++kt) {
    v16h a;
    #pragma unroll
    for (int e = 0; e < 16; ++e) a[e] = arow[kt * 32 + off[e]];
    const h16* bp = Bf + ((size_t)(kt * NT) * 32 + lane) * 16;
    #pragma unroll
    for (int j = 0; j < NT; ++j) {
      v16h bfrag = *(const v16h*)(bp + (size_t)j * 32 * 16);
      acc[j] = __builtin_amdgcn_wmma_f32_16x16x32_f16(
          false, a, false, bfrag, (short)0, acc[j], false, false);
    }
  }

  int ncol = lane & 15;
  int rbase = mtile * 16 + hs * 8;
  #pragma unroll
  for (int j = 0; j < NT; ++j) {
    int col = j * 16 + ncol;
    if (col >= Nc) continue;
    float bv = bias ? bias[col] : 0.f;
    #pragma unroll
    for (int r = 0; r < 8; ++r) {
      int row = rbase + r;
      float v = act_apply(acc[j][r] + bv, act);
      C[(size_t)row * ldc + col] = v;
      if (Ch) Ch[(size_t)row * ldh + col] = (h16)v;
    }
  }
}

// ============================== host side ==================================

static inline int cdiv(long long a, int b) { return (int)((a + b - 1) / b); }

extern "C" void kernel_launch(void* const* d_in, const int* in_sizes, int n_in,
                              void* d_out, int out_size, void* d_ws, size_t ws_size,
                              hipStream_t stream) {
  (void)in_sizes; (void)n_in; (void)out_size; (void)ws_size;

  const float* inp = (const float*)d_in[0];
  const int*   src = (const int*)d_in[1];
  const int*   dst = (const int*)d_in[2];
  const float* We1 = (const float*)d_in[3];
  const float* be1 = (const float*)d_in[4];
  const float* We2 = (const float*)d_in[5];
  const float* be2 = (const float*)d_in[6];
  const float* We3 = (const float*)d_in[7];
  const float* be3 = (const float*)d_in[8];
  const float* Wp1 = (const float*)d_in[9];
  const float* bp1 = (const float*)d_in[10];
  const float* Ws1 = (const float*)d_in[11];
  const float* Wn1 = (const float*)d_in[12];
  const float* b1  = (const float*)d_in[13];
  const float* Wc2 = (const float*)d_in[14];
  const float* bc2 = (const float*)d_in[15];
  const float* Wt3 = (const float*)d_in[16];
  const float* bt3 = (const float*)d_in[17];
  const float* Wp4 = (const float*)d_in[18];
  const float* bp4 = (const float*)d_in[19];
  const float* Ws4 = (const float*)d_in[20];
  const float* Wn4 = (const float*)d_in[21];
  const float* b4  = (const float*)d_in[22];
  const float* Wc5 = (const float*)d_in[23];
  const float* bc5 = (const float*)d_in[24];
  const float* Wt6 = (const float*)d_in[25];
  const float* bt6 = (const float*)d_in[26];
  float* out = (float*)d_out;

  // ---- workspace layout (256B aligned) ----
  size_t off = 0;
  char* ws = (char*)d_ws;
  auto alloc = [&](size_t bytes) -> void* {
    void* p = ws + off;
    off += (bytes + 255) & ~(size_t)255;
    return p;
  };
  h16* h1      = (h16*)alloc((size_t)NNODES * 100 * 98 * 3 * 2);   // 120.4 MB
  h16* h2      = (h16*)alloc((size_t)NNODES * 200 * 96 * 2 * 2);   // 157.3 MB
  h16* B2f     = (h16*)alloc((size_t)19 * 13 * 512 * 2);
  h16* Btmp    = (h16*)alloc((size_t)512 * 128 * 2);
  h16* Wp1f    = (h16*)alloc((size_t)3 * 6 * 512 * 2);
  h16* Bs1f    = (h16*)alloc((size_t)6 * 8 * 512 * 2);
  h16* Wc2f    = (h16*)alloc((size_t)16 * 8 * 512 * 2);
  h16* Wt3f    = (h16*)alloc((size_t)16 * 8 * 512 * 2);
  h16* Wp4f    = (h16*)alloc((size_t)4 * 8 * 512 * 2);
  h16* Bs4f    = (h16*)alloc((size_t)8 * 8 * 512 * 2);
  h16* Wc5f    = (h16*)alloc((size_t)16 * 8 * 512 * 2);
  h16* Wt6f    = (h16*)alloc((size_t)16 * 1 * 512 * 2);
  h16* concat1 = (h16*)alloc((size_t)NNODES * 192 * 2);            // [x96 | hn96]
  h16* concat4 = (h16*)alloc((size_t)NNODES * 256 * 2);            // [x128 | hn128]
  h16* cc      = (h16*)alloc((size_t)NNODES * 512 * 2);            // cheb concat
  h16* tc      = (h16*)alloc((size_t)NNODES * 512 * 2);            // tag concat
  const size_t FB = (size_t)NNODES * 128 * sizeof(float);
  float* xA   = (float*)alloc(FB);
  float* xB   = (float*)alloc(FB);
  float* xC   = (float*)alloc(FB);
  float* xD   = (float*)alloc(FB);
  float* accb = (float*)alloc(FB);
  float* mbuf = (float*)alloc(FB);
  float* hnb  = (float*)alloc(FB);
  float* deg  = (float*)alloc((size_t)NNODES * sizeof(float));
  float* nrm  = (float*)alloc((size_t)NNODES * sizeof(float));

  const int T = 256;

  // ---- encoder ----
  hipMemsetAsync(concat1, 0, (size_t)NNODES * 192 * 2, stream);
  conv1_k<<<cdiv((long long)NNODES * 100 * 98 * 3, T), T, 0, stream>>>(inp, We1, be1, h1);
  pack_We2_k<<<cdiv(19 * 13 * 512, T), T, 0, stream>>>(We2, B2f);
  conv2_wmma_k<<<(NNODES * 192 / 16) / 8, T, 0, stream>>>(h1, B2f, be2, h2);
  conv3_k<<<cdiv(NNODES * 96, T), T, 0, stream>>>(h2, We3, be3, concat1);

  // ---- degree norm ----
  hipMemsetAsync(deg, 0, NNODES * sizeof(float), stream);
  deg_add_k<<<cdiv(NEDGES, T), T, 0, stream>>>(dst, deg);
  norm_k<<<cdiv(NNODES, T), T, 0, stream>>>(deg, nrm);

  // ---- pack graph weights into fragment layout ----
  auto packB = [&](h16* frag, int ktiles, int ntiles,
                   const float* s0, int r0, int rows0,
                   const float* s1, int r1, int rows1, int cols) {
    hipMemsetAsync(Btmp, 0, (size_t)ktiles * 32 * ntiles * 16 * 2, stream);
    int ldb = ntiles * 16;
    copy_pad_k<<<cdiv((long long)rows0 * cols, T), T, 0, stream>>>(s0, rows0, cols, Btmp, ldb, r0);
    if (s1)
      copy_pad_k<<<cdiv((long long)rows1 * cols, T), T, 0, stream>>>(s1, rows1, cols, Btmp, ldb, r1);
    pack_frags_k<<<cdiv(ktiles * ntiles * 512, T), T, 0, stream>>>(Btmp, ldb, ktiles, ntiles, frag);
  };
  packB(Wp1f,  3, 6, Wp1,  0,  94, nullptr, 0,   0,  94);
  packB(Bs1f,  6, 8, Ws1,  0,  94, Wn1,    96,  94, 128);
  packB(Wc2f, 16, 8, Wc2,  0, 512, nullptr, 0,   0, 128);
  packB(Wt3f, 16, 8, Wt3,  0, 512, nullptr, 0,   0, 128);
  packB(Wp4f,  4, 8, Wp4,  0, 128, nullptr, 0,   0, 128);
  packB(Bs4f,  8, 8, Ws4,  0, 128, Wn4,   128, 128, 128);
  packB(Wc5f, 16, 8, Wc5,  0, 512, nullptr, 0,   0, 128);
  packB(Wt6f, 16, 1, Wt6,  0, 512, nullptr, 0,   0,  16);

  const int GB = (NNODES / 16) / 8;  // 16 blocks of 8 waves for M=2048

  auto agg_into = [&](const float* xin, float alpha, const float* z, float beta,
                      float* yout, h16* slot, int ldh) {
    hipMemsetAsync(accb, 0, FB, stream);
    scatter_sum_k<<<cdiv((long long)NEDGES * 128, T), T, 0, stream>>>(xin, src, dst, nrm, accb);
    finalize_k<<<cdiv((long long)NNODES * 128, T), T, 0, stream>>>(accb, nrm, alpha, z, beta,
                                                                   yout, slot, ldh);
  };

  // ---- sage1: x=enc(94) -> x1 (128) ----
  wmma_gemm_k<6><<<GB, T, 0, stream>>>(concat1, 192, 3, Wp1f, bp1, 94,
                                       mbuf, 94, (h16*)nullptr, 0, ACT_RELU);
  hipMemsetAsync(hnb, 0, FB, stream);
  scatter_max_k<<<cdiv((long long)NEDGES * 94, T), T, 0, stream>>>(mbuf, src, dst, hnb, 94);
  cvt_f16_k<<<cdiv((long long)NNODES * 94, T), T, 0, stream>>>(hnb, NNODES, 94, concat1 + 96, 192);
  wmma_gemm_k<8><<<GB, T, 0, stream>>>(concat1, 192, 6, Bs1f, b1, 128,
                                       xA, 128, cc + 0, 512, ACT_TANH2);

  // ---- cheb2: x1 -> x2 ----
  agg_into(xA, -1.f, nullptr, 0.f, xB, cc + 128, 512);        // Tx1
  agg_into(xB, -2.f, xA,     -1.f, xC, cc + 256, 512);        // Tx2
  agg_into(xC, -2.f, xB,     -1.f, xD, cc + 384, 512);        // Tx3
  wmma_gemm_k<8><<<GB, T, 0, stream>>>(cc, 512, 16, Wc2f, bc2, 128,
                                       xA, 128, tc + 0, 512, ACT_TANH);

  // ---- tag3: x2 -> x3 ----
  agg_into(xA, 1.f, nullptr, 0.f, xB, tc + 128, 512);
  agg_into(xB, 1.f, nullptr, 0.f, xC, tc + 256, 512);
  agg_into(xC, 1.f, nullptr, 0.f, xD, tc + 384, 512);
  wmma_gemm_k<8><<<GB, T, 0, stream>>>(tc, 512, 16, Wt3f, bt3, 128,
                                       xA, 128, concat4 + 0, 256, ACT_TANH_LRELU);

  // ---- sage4: x3 -> x4 ----
  wmma_gemm_k<8><<<GB, T, 0, stream>>>(concat4, 256, 4, Wp4f, bp4, 128,
                                       mbuf, 128, (h16*)nullptr, 0, ACT_RELU);
  hipMemsetAsync(hnb, 0, FB, stream);
  scatter_max_k<<<cdiv((long long)NEDGES * 128, T), T, 0, stream>>>(mbuf, src, dst, hnb, 128);
  cvt_f16_k<<<cdiv((long long)NNODES * 128, T), T, 0, stream>>>(hnb, NNODES, 128, concat4 + 128, 256);
  wmma_gemm_k<8><<<GB, T, 0, stream>>>(concat4, 256, 8, Bs4f, b4, 128,
                                       xA, 128, cc + 0, 512, ACT_TANH2);

  // ---- cheb5: x4 -> x5 ----
  agg_into(xA, -1.f, nullptr, 0.f, xB, cc + 128, 512);
  agg_into(xB, -2.f, xA,     -1.f, xC, cc + 256, 512);
  agg_into(xC, -2.f, xB,     -1.f, xD, cc + 384, 512);
  wmma_gemm_k<8><<<GB, T, 0, stream>>>(cc, 512, 16, Wc5f, bc5, 128,
                                       xA, 128, tc + 0, 512, ACT_TANH);

  // ---- tag6: x5 -> sigmoid out [2048,16] ----
  agg_into(xA, 1.f, nullptr, 0.f, xB, tc + 128, 512);
  agg_into(xB, 1.f, nullptr, 0.f, xC, tc + 256, 512);
  agg_into(xC, 1.f, nullptr, 0.f, xD, tc + 384, 512);
  wmma_gemm_k<1><<<GB, T, 0, stream>>>(tc, 512, 16, Wt6f, bt6, 16,
                                       out, 16, (h16*)nullptr, 0, ACT_SIG_LRELU);
}
